// RevisionModel_74741020885425
// MI455X (gfx1250) — compile-verified, matrix operand order checked by reading, not run on hardware
//
#include <hip/hip_runtime.h>

// ===========================================================================
// LSTMN revision model forward for MI455X (gfx1250, wave32).
//
// Persistent-grid implementation: one kernel runs the whole T=256 scan with a
// software device-wide barrier between dependent stages. All matmuls are
// M=16 (batch) x N x K and map 1:1 onto v_wmma_f32_16x16x32_f16 tiles
// (f16 inputs, f32 accumulate). Attention keys (phi_buf @ lk_w^T) are cached
// in a ring buffer and computed incrementally (one new row per step).
//
// GEMM fragments: all K dims forced to multiples of 32 (zero-padded weight
// packs built in the init kernel), loaded as native 4-float vectors
// (global_load_b128) with packed v_cvt_pkrtz f32->f16; non-temporal hints on
// the weight streams (clang ext_vector type, which the builtin accepts).
//
// Input pointer order = setup_inputs() dict insertion order (38 tensors).
// Workspace requirement: ~12.6 MB.
// ===========================================================================

#define DEVINL __device__ __forceinline__

typedef __attribute__((ext_vector_type(16))) _Float16 v16h;
typedef __attribute__((ext_vector_type(8)))  float    v8f;
typedef __attribute__((ext_vector_type(4)))  float    v4f;   // native vec for NT loads
typedef __attribute__((ext_vector_type(2)))  __fp16   fh2;

constexpr int B_ = 16, T_ = 256, E_ = 300, EP_ = 320;   // EP_: K-padded embed
constexpr int HR_ = 512, HC_ = 256, A_ = 256;
constexpr int CACHE_ = 32, LBL_ = 64;

// ---- workspace layout (float offsets) -------------------------------------
constexpr size_t OFF_XEMB = 0;                                  // [T][B][EP]
constexpr size_t OFF_H0   = OFF_XEMB + (size_t)T_ * B_ * EP_;   // [B][HR]
constexpr size_t OFF_C0   = OFF_H0 + (size_t)B_ * HR_;
constexpr size_t OFF_H1   = OFF_C0 + (size_t)B_ * HR_;          // h1 == h_enc
constexpr size_t OFF_C1   = OFF_H1 + (size_t)B_ * HR_;
constexpr size_t OFF_G0   = OFF_C1 + (size_t)B_ * HR_;          // [B][4HR]
constexpr size_t OFF_G1   = OFF_G0 + (size_t)B_ * 4 * HR_;
constexpr size_t OFF_PRED = OFF_G1 + (size_t)B_ * 4 * HR_;      // [B][LBL]
constexpr size_t OFF_Q    = OFF_PRED + (size_t)B_ * LBL_;       // [2][B][A]
constexpr size_t OFF_SS   = OFF_Q + (size_t)2 * B_ * A_;        // [2][B][A]
constexpr size_t OFF_Z    = OFF_SS + (size_t)2 * B_ * A_;       // [B][HR]
constexpr size_t OFF_PHIN = OFF_Z + (size_t)B_ * HR_;           // [B][HC]
constexpr size_t OFF_HS   = OFF_PHIN + (size_t)B_ * HC_;        // h_summ [2][B][HC]
constexpr size_t OFF_CS   = OFF_HS + (size_t)2 * B_ * HC_;      // c_s    [2][B][HC]
constexpr size_t OFF_GATES= OFF_CS + (size_t)2 * B_ * HC_;      // [2][B][4HC]
constexpr size_t OFF_HC0  = OFF_GATES + (size_t)2 * B_ * 4 * HC_;
constexpr size_t OFF_HC1  = OFF_HC0 + (size_t)B_ * HC_;
constexpr size_t OFF_PHIR = OFF_HC1 + (size_t)B_ * HC_;         // phi ring  [32][B][HC]
constexpr size_t OFF_CELLR= OFF_PHIR + (size_t)CACHE_ * B_ * HC_;// cell ring [2][32][B][HC]
constexpr size_t OFF_KKR  = OFF_CELLR + (size_t)2 * CACHE_ * B_ * HC_; // key ring [2][32][B][A]
constexpr size_t OFF_BAR  = OFF_KKR + (size_t)2 * CACHE_ * B_ * A_;    // 2 u32 barrier words
constexpr size_t OFF_WIH0P= OFF_BAR + 8;                        // Wih0 packed [2048][320]
constexpr size_t OFF_WW0XP= OFF_WIH0P + (size_t)4 * HR_ * EP_;  // Ww0 x-part packed [1024][320]
constexpr size_t WS_FLOATS= OFF_WW0XP + (size_t)4 * HC_ * EP_;

struct Pm {
  const int*   x;
  const float* emb;
  const float* Wih0; const float* Whh0; const float* bih0; const float* bhh0;
  const float* Wih1; const float* Whh1; const float* bih1; const float* bhh1;
  const float* proj_w; const float* proj_b;
  const float* Ww0; const float* Wb0; const float* vw0; const float* vb0;
  const float* lqw0; const float* lqb0; const float* lkw0; const float* lkb0;
  const float* lsw0; const float* lsb0;
  const float* Ww1; const float* Wb1; const float* vw1; const float* vb1;
  const float* lqw1; const float* lqb1; const float* lkw1; const float* lkb1;
  const float* lsw1; const float* lsb1;
  const float* z_w; const float* z_b;
  const float* phi_w; const float* phi_b;
  const float* vp_w; const float* vp_b;
};

DEVINL float sigm(float x) { return 1.f / (1.f + __expf(-x)); }

// ---- f32x16 -> f16 fragment with packed-cvt (one 32-K chunk, one lane) ----
// Per ISA 7.12.2: lane k-pattern = two contiguous 8-float runs,
//   run0 @ k0 + khalf*8, run1 @ k0 + 16 + khalf*8  -> 4x 16B vector loads.
union FragU { v16h v; fh2 p[8]; };

DEVINL v16h cvt_frag(v4f x0, v4f x1, v4f x2, v4f x3) {
  FragU u;
  u.p[0] = __builtin_amdgcn_cvt_pkrtz(x0.x, x0.y);
  u.p[1] = __builtin_amdgcn_cvt_pkrtz(x0.z, x0.w);
  u.p[2] = __builtin_amdgcn_cvt_pkrtz(x1.x, x1.y);
  u.p[3] = __builtin_amdgcn_cvt_pkrtz(x1.z, x1.w);
  u.p[4] = __builtin_amdgcn_cvt_pkrtz(x2.x, x2.y);
  u.p[5] = __builtin_amdgcn_cvt_pkrtz(x2.z, x2.w);
  u.p[6] = __builtin_amdgcn_cvt_pkrtz(x3.x, x3.y);
  u.p[7] = __builtin_amdgcn_cvt_pkrtz(x3.z, x3.w);
  return u.v;
}

DEVINL v16h load_frag(const float* __restrict__ row, int k0, int khalf) {
  const float* p0 = row + k0 + khalf * 8;
  return cvt_frag(*(const v4f*)(p0),      *(const v4f*)(p0 + 4),
                  *(const v4f*)(p0 + 16), *(const v4f*)(p0 + 20));
}

DEVINL v16h load_frag_nt(const float* __restrict__ row, int k0, int khalf) {
  const float* p0 = row + k0 + khalf * 8;
  return cvt_frag(__builtin_nontemporal_load((const v4f*)(p0)),
                  __builtin_nontemporal_load((const v4f*)(p0 + 4)),
                  __builtin_nontemporal_load((const v4f*)(p0 + 16)),
                  __builtin_nontemporal_load((const v4f*)(p0 + 20)));
}

// A GEMM source: Out += A[16,K](row-major,lda) * W[N,K](row-major,ldw)^T.
// K must be a multiple of 32; A rows and W rows must be 16B aligned.
struct Src { const float* A; int lda; const float* W; int ldw; int K; };

// One 16x16 output tile per wave. C/D layout: vgpr r, lane l ->
//   M = r + 8*(l>>4), N = l&15  (all 8 C regs of a lane share one N -> free bias).
DEVINL void gemm(int wid, int lane, int base, int count, int ntiles,
                 const Src* src, int nsrc,
                 const float* bias0, const float* bias1,
                 float* Out, int ldo, bool accum, int act)
{
  const int wr = wid - base;
  if (wr < 0 || wr >= count) return;
  const int mrow  = lane & 15;
  const int khalf = lane >> 4;
  for (int tile = wr; tile < ntiles; tile += count) {
    const int ncol = tile * 16 + (lane & 15);
    v8f c;
    if (accum) {
#pragma unroll
      for (int r = 0; r < 8; ++r) c[r] = Out[(size_t)(r + 8 * khalf) * ldo + ncol];
    } else {
      float bv = bias0 ? bias0[ncol] : 0.f;
      if (bias1) bv += bias1[ncol];
#pragma unroll
      for (int r = 0; r < 8; ++r) c[r] = bv;
    }
    for (int s = 0; s < nsrc; ++s) {
      const float* Ar = src[s].A + (size_t)mrow * src[s].lda;
      const float* Wr = src[s].W + (size_t)ncol * src[s].ldw;
      const int K = src[s].K;
      for (int k0 = 0; k0 < K; k0 += 32) {
        const v16h a = load_frag(Ar, k0, khalf);
        const v16h b = load_frag_nt(Wr, k0, khalf);
        c = __builtin_amdgcn_wmma_f32_16x16x32_f16(false, a, false, b,
                                                   (short)0, c, false, false);
      }
    }
#pragma unroll
    for (int r = 0; r < 8; ++r) {
      float v = c[r];
      if (act) v = tanhf(v);
      Out[(size_t)(r + 8 * khalf) * ldo + ncol] = v;
    }
  }
}

// Device-wide two-phase barrier (agent-scope atomics; all 32 WGs resident).
DEVINL void grid_sync(unsigned* cnt, unsigned* gen, unsigned nwg)
{
  __threadfence();          // make this thread's writes agent-visible
  __syncthreads();
  if (threadIdx.x == 0) {
    unsigned g = __hip_atomic_load(gen, __ATOMIC_RELAXED, __HIP_MEMORY_SCOPE_AGENT);
    if (__hip_atomic_fetch_add(cnt, 1u, __ATOMIC_ACQ_REL, __HIP_MEMORY_SCOPE_AGENT) == nwg - 1u) {
      __hip_atomic_store(cnt, 0u, __ATOMIC_RELAXED, __HIP_MEMORY_SCOPE_AGENT);
      __hip_atomic_fetch_add(gen, 1u, __ATOMIC_ACQ_REL, __HIP_MEMORY_SCOPE_AGENT);
    } else {
      while (__hip_atomic_load(gen, __ATOMIC_ACQUIRE, __HIP_MEMORY_SCOPE_AGENT) == g)
        __builtin_amdgcn_s_sleep(1);
    }
  }
  __syncthreads();
  __threadfence();          // acquire: invalidate stale near caches
}

__global__ void init_state(Pm p, float* __restrict__ ws)
{
  const size_t tid = (size_t)blockIdx.x * blockDim.x + threadIdx.x;
  const size_t nth = (size_t)gridDim.x * blockDim.x;
  // embed all tokens, K-padded to EP_ with zeros (removes gather from scan)
  for (size_t idx = tid; idx < (size_t)T_ * B_ * EP_; idx += nth) {
    const int e = (int)(idx % EP_);
    const size_t tb = idx / EP_;
    const int b = (int)(tb % B_);
    const int tt = (int)(tb / B_);
    ws[OFF_XEMB + idx] = (e < E_) ? p.emb[(size_t)p.x[(size_t)b * T_ + tt] * E_ + e] : 0.f;
  }
  // zero-padded weight packs so every GEMM K is a multiple of 32
  for (size_t idx = tid; idx < (size_t)4 * HR_ * EP_; idx += nth) {   // Wih0 [2048][320]
    const int k = (int)(idx % EP_);
    const size_t n = idx / EP_;
    ws[OFF_WIH0P + idx] = (k < E_) ? p.Wih0[n * E_ + k] : 0.f;
  }
  for (size_t idx = tid; idx < (size_t)4 * HC_ * EP_; idx += nth) {   // Ww0[:,256:556] [1024][320]
    const int k = (int)(idx % EP_);
    const size_t n = idx / EP_;
    ws[OFF_WW0XP + idx] = (k < E_) ? p.Ww0[n * (HC_ + E_) + HC_ + k] : 0.f;
  }
  for (size_t i = tid; i < (size_t)4 * B_ * HR_; i += nth) ws[OFF_H0 + i] = 0.f; // h0,c0,h1,c1
  for (size_t i = tid; i < (size_t)2 * B_ * HC_; i += nth) ws[OFF_HS + i] = 0.f; // h_summ
  if (tid == 0) {
    unsigned* bar = (unsigned*)(ws + OFF_BAR);
    bar[0] = 0u; bar[1] = 0u;
  }
}

__global__ __launch_bounds__(256, 1) void lstmn_forward(Pm p, float* __restrict__ ws,
                                                        float* __restrict__ out_preds,
                                                        float* __restrict__ out_pols)
{
  const int tidb = threadIdx.x;
  const int tid  = blockIdx.x * 256 + tidb;
  const int nth  = gridDim.x * 256;          // 8192 threads
  const int lane = tidb & 31;
  const int wid  = tid >> 5;                 // 0..255 waves
  const int wv   = tidb >> 5;

  float* xemb = ws + OFF_XEMB;
  float* h0 = ws + OFF_H0;  float* c0 = ws + OFF_C0;
  float* h1 = ws + OFF_H1;  float* c1 = ws + OFF_C1;
  float* G0 = ws + OFF_G0;  float* G1 = ws + OFF_G1;
  float* pred = ws + OFF_PRED;
  float* qb = ws + OFF_Q;   float* ssb = ws + OFF_SS;
  float* zb = ws + OFF_Z;   float* phin = ws + OFF_PHIN;
  float* hsum = ws + OFF_HS; float* csb = ws + OFF_CS;
  float* gates = ws + OFF_GATES;
  float* hc0 = ws + OFF_HC0; float* hc1 = ws + OFF_HC1;
  float* phir = ws + OFF_PHIR; float* cellr = ws + OFF_CELLR; float* kkr = ws + OFF_KKR;
  const float* Wih0p = ws + OFF_WIH0P;
  const float* Ww0xp = ws + OFF_WW0XP;
  unsigned* cnt = (unsigned*)(ws + OFF_BAR);
  unsigned* gen = cnt + 1;
  const unsigned NWG = gridDim.x;

  const float* lkw[2] = {p.lkw0, p.lkw1};  const float* lkb[2] = {p.lkb0, p.lkb1};
  const float* lsw[2] = {p.lsw0, p.lsw1};  const float* lsb[2] = {p.lsb0, p.lsb1};
  const float* lqw[2] = {p.lqw0, p.lqw1};  const float* lqb[2] = {p.lqb0, p.lqb1};
  const float* vw[2]  = {p.vw0,  p.vw1};   const float* vb[2]  = {p.vb0,  p.vb1};

  __shared__ float probS[8][32];

  for (int t = 0; t < T_; ++t) {
    const float* xe = xemb + (size_t)t * B_ * EP_;
    const int slot = t & (CACHE_ - 1);

    // ---- Stage A: G0; gates0 x-part; ss (prev h_s); new attention keys ----
    {
      Src s2[2] = { {xe, EP_, Wih0p, EP_, EP_}, {h0, HR_, p.Whh0, HR_, HR_} };
      gemm(wid, lane, 0, 128, 128, s2, 2, p.bih0, p.bhh0, G0, 4 * HR_, false, 0);
      Src sx[1] = { {xe, EP_, Ww0xp, EP_, EP_} };
      gemm(wid, lane, 128, 64, 64, sx, 1, p.Wb0, nullptr, gates, 4 * HC_, false, 0);
      for (int i = 0; i < 2; ++i) {
        Src s1[1] = { {hsum + (size_t)i * B_ * A_, A_, lsw[i], HC_, HC_} };
        gemm(wid, lane, 192 + 16 * i, 16, 16, s1, 1, lsb[i], nullptr,
             ssb + (size_t)i * B_ * A_, A_, false, 0);
      }
      if (t > 0) {
        const int ps = (t - 1) & (CACHE_ - 1);
        for (int i = 0; i < 2; ++i) {
          Src sk[1] = { {phin, HC_, lkw[i], HC_, HC_} };
          gemm(wid, lane, 224 + 16 * i, 16, 16, sk, 1, lkb[i], nullptr,
               kkr + ((size_t)(i * CACHE_ + ps) * B_) * A_, A_, false, 0);
        }
      }
    }
    grid_sync(cnt, gen, NWG);

    // ---- Stage B: LSTM layer 0 update (gate order i,f,g,o) ----
    for (int e = tid; e < B_ * HR_; e += nth) {
      const int b = e >> 9, j = e & (HR_ - 1);
      const float* g = G0 + (size_t)b * 4 * HR_;
      const float cn = sigm(g[HR_ + j]) * c0[e] + sigm(g[j]) * tanhf(g[2 * HR_ + j]);
      c0[e] = cn;
      h0[e] = sigm(g[3 * HR_ + j]) * tanhf(cn);
    }
    grid_sync(cnt, gen, NWG);

    // ---- Stage C: G1 ----
    {
      Src s2[2] = { {h0, HR_, p.Wih1, HR_, HR_}, {h1, HR_, p.Whh1, HR_, HR_} };
      gemm(wid, lane, 0, 128, 128, s2, 2, p.bih1, p.bhh1, G1, 4 * HR_, false, 0);
    }
    grid_sync(cnt, gen, NWG);

    // ---- Stage D: LSTM layer 1 update -> h_enc ----
    for (int e = tid; e < B_ * HR_; e += nth) {
      const int b = e >> 9, j = e & (HR_ - 1);
      const float* g = G1 + (size_t)b * 4 * HR_;
      const float cn = sigm(g[HR_ + j]) * c1[e] + sigm(g[j]) * tanhf(g[2 * HR_ + j]);
      c1[e] = cn;
      h1[e] = sigm(g[3 * HR_ + j]) * tanhf(cn);
    }
    grid_sync(cnt, gen, NWG);

    // ---- Stage E: pred, q0, q1 ----
    {
      Src sp[1] = { {h1, HR_, p.proj_w, HR_, HR_} };
      gemm(wid, lane, 0, 4, 4, sp, 1, p.proj_b, nullptr, pred, LBL_, false, 0);
      for (int i = 0; i < 2; ++i) {
        Src sq[1] = { {h1, HR_, lqw[i], HR_, HR_} };
        gemm(wid, lane, 4 + 16 * i, 16, 16, sq, 1, lqb[i], nullptr,
             qb + (size_t)i * B_ * A_, A_, false, 0);
      }
    }
    grid_sync(cnt, gen, NWG);

    // ---- Stage F: attention (both ctrls), z = tanh(pred@z_w^T+b), preds out ----
    if (wid < 32) {                       // one wave per (ctrl,batch)
      const int i = wid >> 4, b = wid & 15;
      float prob = 0.f;
      if (t > 0) {                        // lane == cache slot; valid iff slot written
        const bool valid = lane < t;
        float acc = 0.f;
        const float* qq = qb  + (size_t)(i * B_ + b) * A_;
        const float* sv = ssb + (size_t)(i * B_ + b) * A_;
        const float* kv = kkr + ((size_t)(i * CACHE_ + lane) * B_ + b) * A_;
        const float* vwi = vw[i];
        for (int a2 = 0; a2 < A_; ++a2)
          acc += tanhf(qq[a2] + sv[a2] + kv[a2]) * vwi[a2];
        float sc = valid ? (acc + vb[i][0]) : -3.0e38f;
        float m = sc;
        for (int d = 16; d; d >>= 1) m = fmaxf(m, __shfl_xor(m, d, 32));
        float ex = valid ? __expf(sc - m) : 0.f;
        float sm = ex;
        for (int d = 16; d; d >>= 1) sm += __shfl_xor(sm, d, 32);
        prob = ex / sm;
      }
      probS[wv][lane] = prob;
      const int lim = (t < CACHE_) ? t : CACHE_;
      for (int j = lane; j < HC_; j += 32) {
        float ha = 0.f, ca = 0.f;
        for (int l2 = 0; l2 < lim; ++l2) {
          const float pl = probS[wv][l2];
          ha += pl * phir[((size_t)l2 * B_ + b) * HC_ + j];
          ca += pl * cellr[((size_t)(i * CACHE_ + l2) * B_ + b) * HC_ + j];
        }
        hsum[(size_t)(i * B_ + b) * HC_ + j] = ha;   // == 0 when t==0 (live=0)
        csb [(size_t)(i * B_ + b) * HC_ + j] = ca;
      }
    }
    {
      Src sz[1] = { {pred, LBL_, p.z_w, LBL_, LBL_} };
      gemm(wid, lane, 32, 32, 32, sz, 1, p.z_b, nullptr, zb, HR_, false, 1);
    }
    for (int e = tid; e < B_ * LBL_; e += nth) {     // preds: [B,T,LBL]
      const int b = e >> 6, n = e & 63;
      out_preds[((size_t)b * T_ + t) * LBL_ + n] = pred[e];
    }
    grid_sync(cnt, gen, NWG);

    // ---- Stage G: gates0 += h_s0 part; phi = tanh([h_enc,z]@phi_w^T+b) ----
    {
      Src sg[1] = { {hsum, A_, p.Ww0, HC_ + E_, HC_} };
      gemm(wid, lane, 0, 64, 64, sg, 1, nullptr, nullptr, gates, 4 * HC_, true, 0);
      Src sp2[2] = { {h1, HR_, p.phi_w, 2 * HR_, HR_},
                     {zb, HR_, p.phi_w + HR_, 2 * HR_, HR_} };
      gemm(wid, lane, 64, 16, 16, sp2, 2, p.phi_b, nullptr, phin, HC_, false, 1);
    }
    grid_sync(cnt, gen, NWG);

    // ---- Stage H: ctrl0 update (gate slice order f,o,i,ch) ----
    for (int e = tid; e < B_ * HC_; e += nth) {
      const int b = e >> 8, a = e & 255;
      const float* g = gates + (size_t)b * 4 * HC_;
      const float cc = sigm(g[a]) * csb[e] + sigm(g[2 * HC_ + a]) * tanhf(g[3 * HC_ + a]);
      cellr[((size_t)slot * B_ + b) * HC_ + a] = cc;            // ctrl0 ring
      hc0[e] = sigm(g[HC_ + a]) * tanhf(cc);
    }
    grid_sync(cnt, gen, NWG);

    // ---- Stage I: gates1 = [h_s1, h_ctrl0] @ Ww1^T + Wb1 ----
    {
      Src sg[2] = { {hsum + (size_t)B_ * A_, A_, p.Ww1, 2 * HC_, HC_},
                    {hc0, HC_, p.Ww1 + HC_, 2 * HC_, HC_} };
      gemm(wid, lane, 0, 64, 64, sg, 2, p.Wb1, nullptr,
           gates + (size_t)B_ * 4 * HC_, 4 * HC_, false, 0);
    }
    grid_sync(cnt, gen, NWG);

    // ---- Stage J: ctrl1 update ----
    for (int e = tid; e < B_ * HC_; e += nth) {
      const int b = e >> 8, a = e & 255;
      const float* g = gates + (size_t)(B_ + b) * 4 * HC_;
      const float cc = sigm(g[a]) * csb[(size_t)B_ * HC_ + e]
                     + sigm(g[2 * HC_ + a]) * tanhf(g[3 * HC_ + a]);
      cellr[((size_t)(CACHE_ + slot) * B_ + b) * HC_ + a] = cc; // ctrl1 ring
      hc1[e] = sigm(g[HC_ + a]) * tanhf(cc);
    }
    grid_sync(cnt, gen, NWG);

    // ---- Stage K: policy; append phi to ring ----
    if (wid < B_) {
      const int b = wid;
      float acc = 0.f;
      for (int j = lane; j < HC_; j += 32) acc += hc1[(size_t)b * HC_ + j] * p.vp_w[j];
      for (int d = 16; d; d >>= 1) acc += __shfl_xor(acc, d, 32);
      if (lane == 0) out_pols[(size_t)b * T_ + t] = sigm(acc + p.vp_b[0]);
    }
    for (int e = tid; e < B_ * HC_; e += nth)
      phir[(size_t)slot * B_ * HC_ + e] = phin[e];
    grid_sync(cnt, gen, NWG);
  }
}

extern "C" void kernel_launch(void* const* d_in, const int* in_sizes, int n_in,
                              void* d_out, int out_size, void* d_ws, size_t ws_size,
                              hipStream_t stream)
{
  (void)in_sizes; (void)n_in; (void)out_size; (void)ws_size;
  int i = 0;
  Pm p;
  p.x      = (const int*)d_in[i++];
  p.emb    = (const float*)d_in[i++];
  p.Wih0   = (const float*)d_in[i++];
  p.Whh0   = (const float*)d_in[i++];
  p.bih0   = (const float*)d_in[i++];
  p.bhh0   = (const float*)d_in[i++];
  p.Wih1   = (const float*)d_in[i++];
  p.Whh1   = (const float*)d_in[i++];
  p.bih1   = (const float*)d_in[i++];
  p.bhh1   = (const float*)d_in[i++];
  p.proj_w = (const float*)d_in[i++];
  p.proj_b = (const float*)d_in[i++];
  p.Ww0    = (const float*)d_in[i++];
  p.Wb0    = (const float*)d_in[i++];
  p.vw0    = (const float*)d_in[i++];
  p.vb0    = (const float*)d_in[i++];
  p.lqw0   = (const float*)d_in[i++];
  p.lqb0   = (const float*)d_in[i++];
  p.lkw0   = (const float*)d_in[i++];
  p.lkb0   = (const float*)d_in[i++];
  p.lsw0   = (const float*)d_in[i++];
  p.lsb0   = (const float*)d_in[i++];
  p.Ww1    = (const float*)d_in[i++];
  p.Wb1    = (const float*)d_in[i++];
  p.vw1    = (const float*)d_in[i++];
  p.vb1    = (const float*)d_in[i++];
  p.lqw1   = (const float*)d_in[i++];
  p.lqb1   = (const float*)d_in[i++];
  p.lkw1   = (const float*)d_in[i++];
  p.lkb1   = (const float*)d_in[i++];
  p.lsw1   = (const float*)d_in[i++];
  p.lsb1   = (const float*)d_in[i++];
  p.z_w    = (const float*)d_in[i++];
  p.z_b    = (const float*)d_in[i++];
  p.phi_w  = (const float*)d_in[i++];
  p.phi_b  = (const float*)d_in[i++];
  p.vp_w   = (const float*)d_in[i++];
  p.vp_b   = (const float*)d_in[i++];

  float* ws   = (float*)d_ws;
  float* outp = (float*)d_out;

  init_state<<<dim3(1024), dim3(256), 0, stream>>>(p, ws);
  lstmn_forward<<<dim3(32), dim3(256), 0, stream>>>(p, ws, outp,
                                                    outp + (size_t)B_ * T_ * LBL_);
}
